// MatformerEquivariant_18313740550211
// MI455X (gfx1250) — compile-verified, matrix operand order checked by reading, not run on hardware
//
#include <hip/hip_runtime.h>
#include <hip/hip_bf16.h>
#include <math.h>

#define NNODES 2048
#define NEDGES 16384
#define NGRAPH 32
#define DIN    92
#define DMODEL 256
#define H3     768
#define NSDIM  64
#define NVDIM  8
#define WNUMD  5120   // 64*64 + 2*64*8
#define RBINS  256
#define ECHUNK 2048   // 2048*5120*4B = 42MB wchunk -> stays resident in 192MB L2

typedef __attribute__((ext_vector_type(16))) __bf16 v16bf;
typedef __attribute__((ext_vector_type(8)))  float  v8f;
typedef unsigned short u16;
typedef unsigned int   u32;
typedef __attribute__((ext_vector_type(4))) u32 u32x4;

__device__ __forceinline__ u16 f2bf(float f){
  u32 u = __builtin_bit_cast(u32, f);
  u32 r = u + 0x7FFFu + ((u >> 16) & 1u);   // round-to-nearest-even
  return (u16)(r >> 16);
}

__device__ __forceinline__ float activ(float v, int act){
  if (act == 1) return v > 0.f ? v : 0.f;                    // relu
  if (act == 2) return v > 20.f ? v : log1pf(__expf(v));     // softplus
  if (act == 3) return v / (1.f + __expf(-v));               // silu
  return v;
}

union FragBF { v16bf v; u32 u[8]; };
union Pack4  { u32x4 v; u16 h[8]; };

// ---------------------------------------------------------------------------
// FULL-TILE bf16 WMMA GEMM (M%128==0, K%32==0, N%64==0).
// Block = 128 threads (4 waves); block tile 128x64; wave tile 32x64.
// LDS double-buffered: next K-tile's b128 global loads are issued before the
// current tile's 8 WMMAs, overlapping VMEM with matrix ops via loadcnt.
// Per K-step per wave: 2 A frags + 4 B frags (48 LDS dwords) -> 8 WMMA.
// ---------------------------------------------------------------------------
__global__ __launch_bounds__(128) void k_wmma_gemm_ft(
    const u16* __restrict__ A, const u16* __restrict__ B,
    const float* __restrict__ bias, float* __restrict__ Cf,
    u16* __restrict__ Cbf, int M, int K, int N, int act)
{
  __shared__ __align__(16) u16 As[2][128][40];  // 80B row stride (16B aligned)
  __shared__ __align__(16) u16 Bs[2][64][34];   // transposed: Bs[n][k]
  const int m0 = blockIdx.y * 128, n0 = blockIdx.x * 64;
  const int t = threadIdx.x;
  const int lane = t & 31, wv = t >> 5;
  const int r = lane & 15, hi = lane >> 4;

  // per-thread staging coordinates
  const int arow[4] = { (t + 0) >> 2, (t + 128) >> 2, (t + 256) >> 2, (t + 384) >> 2 };
  const int aseg    = t & 3;                 // same for all i since 128%4==0
  const int bk [2]  = { (t + 0) >> 3, (t + 128) >> 3 };
  const int bns     = t & 7;

  v8f zero = {0.f,0.f,0.f,0.f,0.f,0.f,0.f,0.f};
  v8f acc0[4], acc1[4];
  #pragma unroll
  for (int i = 0; i < 4; ++i){ acc0[i] = zero; acc1[i] = zero; }

  const int nk = K >> 5;
  u32x4 ar[4], br[2];

  // prologue: stage tile 0
  #pragma unroll
  for (int i = 0; i < 4; ++i)
    ar[i] = *(const u32x4*)(A + (size_t)(m0 + arow[i])*K + aseg*8);
  #pragma unroll
  for (int i = 0; i < 2; ++i)
    br[i] = *(const u32x4*)(B + (size_t)bk[i]*N + n0 + bns*8);
  #pragma unroll
  for (int i = 0; i < 4; ++i) *(u32x4*)&As[0][arow[i]][aseg*8] = ar[i];
  #pragma unroll
  for (int i = 0; i < 2; ++i){
    Pack4 p; p.v = br[i];
    #pragma unroll
    for (int j = 0; j < 8; ++j) Bs[0][bns*8 + j][bk[i]] = p.h[j];
  }
  __syncthreads();

  for (int kt = 0; kt < nk; ++kt){
    const int buf = kt & 1;
    const bool more = (kt + 1 < nk);
    if (more){                       // issue next tile's global loads now
      const int k0 = (kt + 1) << 5;
      #pragma unroll
      for (int i = 0; i < 4; ++i)
        ar[i] = *(const u32x4*)(A + (size_t)(m0 + arow[i])*K + k0 + aseg*8);
      #pragma unroll
      for (int i = 0; i < 2; ++i)
        br[i] = *(const u32x4*)(B + (size_t)(k0 + bk[i])*N + n0 + bns*8);
    }
    // compute current tile from LDS
    FragBF a0, a1;
    const u16* ap0 = &As[buf][32*wv +      r][0];
    const u16* ap1 = &As[buf][32*wv + 16 + r][0];
    #pragma unroll
    for (int j = 0; j < 8; ++j){
      int kk = 16*(j >> 2) + 8*hi + 2*(j & 3);
      a0.u[j] = *(const u32*)(ap0 + kk);
      a1.u[j] = *(const u32*)(ap1 + kk);
    }
    #pragma unroll
    for (int jt = 0; jt < 4; ++jt){
      FragBF b;
      const u16* bp = &Bs[buf][16*jt + r][0];
      #pragma unroll
      for (int j = 0; j < 8; ++j){
        int kk = 16*(j >> 2) + 8*hi + 2*(j & 3);
        b.u[j] = *(const u32*)(bp + kk);
      }
      acc0[jt] = __builtin_amdgcn_wmma_f32_16x16x32_bf16(
          false, a0.v, false, b.v, (short)0, acc0[jt], false, false);
      acc1[jt] = __builtin_amdgcn_wmma_f32_16x16x32_bf16(
          false, a1.v, false, b.v, (short)0, acc1[jt], false, false);
    }
    if (more){
      __syncthreads();               // all waves done reading buf^1
      const int nb = buf ^ 1;
      #pragma unroll
      for (int i = 0; i < 4; ++i) *(u32x4*)&As[nb][arow[i]][aseg*8] = ar[i];
      #pragma unroll
      for (int i = 0; i < 2; ++i){
        Pack4 p; p.v = br[i];
        #pragma unroll
        for (int j = 0; j < 8; ++j) Bs[nb][bns*8 + j][bk[i]] = p.h[j];
      }
      __syncthreads();
    }
  }
  // epilogue (no guards: full tile)
  #pragma unroll
  for (int half = 0; half < 2; ++half){
    v8f* ac = half ? acc1 : acc0;
    #pragma unroll
    for (int jt = 0; jt < 4; ++jt){
      #pragma unroll
      for (int rr = 0; rr < 8; ++rr){
        int m = m0 + 32*wv + 16*half + rr + 8*hi;
        int n = n0 + 16*jt + r;
        float v = ac[jt][rr];
        if (bias) v += bias[n];
        v = activ(v, act);
        if (Cf)  Cf[(size_t)m*N + n] = v;
        if (Cbf) Cbf[(size_t)m*N + n] = f2bf(v);
      }
    }
  }
}

// ---------------------------------------------------------------------------
// Guarded bf16 WMMA GEMM (ragged M / K). Used only for embed (K=92) and the
// 32-row readout GEMM. Block = 128 threads; 64x64 tile.
// ---------------------------------------------------------------------------
__global__ __launch_bounds__(128) void k_wmma_gemm_g(
    const u16* __restrict__ A, const u16* __restrict__ B,
    const float* __restrict__ bias, float* __restrict__ Cf,
    u16* __restrict__ Cbf, int M, int K, int N, int act)
{
  __shared__ u16 As[64][34];
  __shared__ u16 Bs[64][34];
  const int m0 = blockIdx.y * 64, n0 = blockIdx.x * 64;
  const int t = threadIdx.x;
  const int lane = t & 31, wv = t >> 5;
  const int r = lane & 15, hi = lane >> 4;

  v8f zero = {0.f,0.f,0.f,0.f,0.f,0.f,0.f,0.f};
  v8f acc[4]; acc[0]=zero; acc[1]=zero; acc[2]=zero; acc[3]=zero;

  for (int k0 = 0; k0 < K; k0 += 32) {
    #pragma unroll
    for (int i = 0; i < 8; ++i) {
      int p = t + 128*i;
      int ar = p >> 4, kk = (p & 15) << 1;
      u32 val = 0;
      if (m0 + ar < M && k0 + kk < K)       // K always even
        val = *(const u32*)(A + (size_t)(m0 + ar)*K + k0 + kk);
      *(u32*)&As[ar][kk] = val;
    }
    #pragma unroll
    for (int i = 0; i < 16; ++i) {
      int p = t + 128*i;
      int nn = p & 63, kk = p >> 6;
      u16 val = 0;
      if (k0 + kk < K) val = B[(size_t)(k0 + kk)*N + n0 + nn];
      Bs[nn][kk] = val;
    }
    if (k0 + 32 < K) {   // CDNA5 global_prefetch_b8 for next K tile
      __builtin_prefetch(A + (size_t)(m0 + (t >> 1))*K + k0 + 32, 0, 1);
      __builtin_prefetch(B + (size_t)(k0 + 32 + lane)*N + n0, 0, 1);
    }
    __syncthreads();
    FragBF a;
    const u16* arow = &As[16*wv + r][0];
    #pragma unroll
    for (int j = 0; j < 8; ++j) {
      int kk = 16*(j >> 2) + 8*hi + 2*(j & 3);
      a.u[j] = *(const u32*)(arow + kk);
    }
    #pragma unroll
    for (int jt = 0; jt < 4; ++jt) {
      FragBF b;
      const u16* brow = &Bs[16*jt + r][0];
      #pragma unroll
      for (int j = 0; j < 8; ++j) {
        int kk = 16*(j >> 2) + 8*hi + 2*(j & 3);
        b.u[j] = *(const u32*)(brow + kk);
      }
      acc[jt] = __builtin_amdgcn_wmma_f32_16x16x32_bf16(
          false, a.v, false, b.v, (short)0, acc[jt], false, false);
    }
    __syncthreads();
  }
  #pragma unroll
  for (int jt = 0; jt < 4; ++jt) {
    #pragma unroll
    for (int rr = 0; rr < 8; ++rr) {
      int m = m0 + 16*wv + rr + 8*hi;
      int n = n0 + 16*jt + r;
      if (m < M) {
        float v = acc[jt][rr];
        if (bias) v += bias[n];
        v = activ(v, act);
        if (Cf)  Cf[(size_t)m*N + n] = v;
        if (Cbf) Cbf[(size_t)m*N + n] = f2bf(v);
      }
    }
  }
}

// ---------------------------------------------------------------------------
// Elementwise / graph kernels
// ---------------------------------------------------------------------------
__global__ void k_f32_to_bf16(const float* __restrict__ s, u16* __restrict__ d, int n){
  for (int i = blockIdx.x*blockDim.x + threadIdx.x; i < n; i += gridDim.x*blockDim.x)
    d[i] = f2bf(s[i]);
}

__global__ void k_mul_bf(const float* __restrict__ a, const float* __restrict__ g,
                         u16* __restrict__ o, int n){
  for (int i = blockIdx.x*blockDim.x + threadIdx.x; i < n; i += gridDim.x*blockDim.x)
    o[i] = f2bf(a[i] * g[i]);
}

__global__ void k_add(float* __restrict__ h, const float* __restrict__ t,
                      u16* __restrict__ hbf, int n){
  for (int i = blockIdx.x*blockDim.x + threadIdx.x; i < n; i += gridDim.x*blockDim.x){
    float v = h[i] + t[i]; h[i] = v; hbf[i] = f2bf(v);
  }
}

__global__ __launch_bounds__(256) void k_rbf(const float* __restrict__ ea, u16* __restrict__ o){
  int e = blockIdx.x, t = threadIdx.x;
  float x = ea[e*3+0], y = ea[e*3+1], z = ea[e*3+2];
  float d = -0.75f * rsqrtf(x*x + y*y + z*z);
  float c = -4.f + (4.f/255.f)*(float)t;
  float v = __expf(-63.75f*(d - c)*(d - c));
  o[(size_t)e*RBINS + t] = f2bf(v);
}

__global__ void k_sph(const float* __restrict__ ea, float* __restrict__ sh1,
                      float* __restrict__ sh2){
  for (int e = blockIdx.x*blockDim.x + threadIdx.x; e < NEDGES; e += gridDim.x*blockDim.x){
    float x = ea[e*3+0], y = ea[e*3+1], z = ea[e*3+2];
    float in = rsqrtf(x*x + y*y + z*z);
    x *= in; y *= in; z *= in;
    const float s3 = 1.7320508f, s15 = 3.8729833f, s5 = 2.2360680f;
    sh1[e*3+0] = s3*x; sh1[e*3+1] = s3*y; sh1[e*3+2] = s3*z;
    sh2[e*5+0] = s15*x*y;
    sh2[e*5+1] = s15*y*z;
    sh2[e*5+2] = 0.5f*s5*(3.f*z*z - 1.f);
    sh2[e*5+3] = s15*x*z;
    sh2[e*5+4] = 0.5f*s15*(x*x - y*y);
  }
}

__global__ __launch_bounds__(256) void k_build_key_alpha(
    const float* __restrict__ qv, const float* __restrict__ kv,
    const float* __restrict__ em, const int* __restrict__ rowI,
    const int* __restrict__ colI, float* __restrict__ alpha){
  int e = blockIdx.x, t = threadIdx.x;
  int c = colI[e], r = rowI[e];
  float q = qv[(size_t)c*DMODEL + t] * 0.0360843918f;   // 1/sqrt(3*256)
  float* a = alpha + (size_t)e*H3;
  a[t]       = q * kv[(size_t)c*DMODEL + t];
  a[256 + t] = q * kv[(size_t)r*DMODEL + t];
  a[512 + t] = q * em[(size_t)e*DMODEL + t];
}

__global__ __launch_bounds__(256) void k_build_msg(
    const float* __restrict__ vv, const float* __restrict__ em,
    const int* __restrict__ rowI, const int* __restrict__ colI,
    u16* __restrict__ m){
  int e = blockIdx.x, t = threadIdx.x;
  int c = colI[e], r = rowI[e];
  u16* o = m + (size_t)e*H3;
  o[t]       = f2bf(vv[(size_t)c*DMODEL + t]);
  o[256 + t] = f2bf(vv[(size_t)r*DMODEL + t]);
  o[512 + t] = f2bf(em[(size_t)e*DMODEL + t]);
}

__global__ __launch_bounds__(256) void k_layernorm(float* __restrict__ x,
    const float* __restrict__ g, const float* __restrict__ b, int width, int post){
  __shared__ float red[256];
  float* xr = x + (size_t)blockIdx.x*width;
  int t = threadIdx.x;
  float s = 0.f;
  for (int j = t; j < width; j += 256) s += xr[j];
  red[t] = s; __syncthreads();
  for (int o = 128; o > 0; o >>= 1){ if (t < o) red[t] += red[t+o]; __syncthreads(); }
  float mu = red[0] / (float)width; __syncthreads();
  float q = 0.f;
  for (int j = t; j < width; j += 256){ float d = xr[j] - mu; q += d*d; }
  red[t] = q; __syncthreads();
  for (int o = 128; o > 0; o >>= 1){ if (t < o) red[t] += red[t+o]; __syncthreads(); }
  float rinv = rsqrtf(red[0] / (float)width + 1e-5f); __syncthreads();
  for (int j = t; j < width; j += 256){
    float y = (xr[j] - mu)*rinv*g[j] + b[j];
    if (post == 1) y = 1.f/(1.f + __expf(-y));
    xr[j] = y;
  }
}

__global__ __launch_bounds__(256) void k_scatter_add(const float* __restrict__ src,
    const int* __restrict__ idx, float* __restrict__ dst, int width){
  int e = blockIdx.x, t = threadIdx.x;
  atomicAdd(&dst[(size_t)idx[e]*width + t], src[(size_t)e*width + t]);
}

__global__ void k_count(const int* __restrict__ idx, int n, float* __restrict__ cnt){
  for (int i = blockIdx.x*blockDim.x + threadIdx.x; i < n; i += gridDim.x*blockDim.x)
    atomicAdd(&cnt[idx[i]], 1.f);
}

__global__ void k_divmean(float* __restrict__ v, const float* __restrict__ cnt,
                          int n, int width){
  int tot = n * width;
  for (int i = blockIdx.x*blockDim.x + threadIdx.x; i < tot; i += gridDim.x*blockDim.x)
    v[i] /= fmaxf(cnt[i / width], 1.f);
}

__global__ __launch_bounds__(256) void k_colstats(const float* __restrict__ x,
                                                  int rows, float* __restrict__ st){
  __shared__ float rs[256], rq[256];
  int c = blockIdx.x, t = threadIdx.x;
  float s = 0.f, q = 0.f;
  for (int r = t; r < rows; r += 256){ float v = x[(size_t)r*DMODEL + c]; s += v; q += v*v; }
  rs[t] = s; rq[t] = q; __syncthreads();
  for (int o = 128; o > 0; o >>= 1){
    if (t < o){ rs[t] += rs[t+o]; rq[t] += rq[t+o]; } __syncthreads();
  }
  if (t == 0){
    float mu = rs[0] / rows;
    float var = rq[0] / rows - mu*mu;
    st[c] = mu; st[DMODEL + c] = rsqrtf(var + 1e-5f);
  }
}

__global__ __launch_bounds__(256) void k_bn_silu(const float* __restrict__ x,
    const float* __restrict__ st, const float* __restrict__ g, const float* __restrict__ b,
    float* __restrict__ hf, u16* __restrict__ hbf){
  int row = blockIdx.x, t = threadIdx.x;
  size_t i = (size_t)row*DMODEL + t;
  float v = (x[i] - st[t]) * st[DMODEL + t] * g[t] + b[t];
  float y = v / (1.f + __expf(-v));
  hf[i] = y; hbf[i] = f2bf(y);
}

__global__ __launch_bounds__(64) void k_equi1(const float* __restrict__ w,
    const float* __restrict__ s, const int* __restrict__ srcI, const int* __restrict__ dstI,
    const float* __restrict__ sh1, const float* __restrict__ sh2,
    float* __restrict__ n0s, float* __restrict__ n1s, float* __restrict__ n2s, int ebase){
  int e = ebase + blockIdx.x, t = threadIdx.x;
  __shared__ float se[64], c1s[8], c2s[8], s1[3], s2[5];
  int sn = srcI[e], dn = dstI[e];
  se[t] = s[(size_t)dn*NSDIM + t];
  if (t < 3) s1[t] = sh1[(size_t)e*3 + t];
  if (t < 5) s2[t] = sh2[(size_t)e*5 + t];
  __syncthreads();
  const float* we = w + (size_t)blockIdx.x*WNUMD;
  const float inv1 = 0.125f;  // 1/sqrt(64)
  float m0 = 0.f;
  for (int u = 0; u < 64; ++u) m0 += we[u*64 + t] * se[u];
  atomicAdd(&n0s[(size_t)sn*NSDIM + t], inv1*m0);
  if (t < 8){
    float c1 = 0.f, c2 = 0.f;
    for (int u = 0; u < 64; ++u){
      float sv = se[u];
      c1 += we[4096 + u*8 + t] * sv;
      c2 += we[4608 + u*8 + t] * sv;
    }
    c1s[t] = inv1*c1; c2s[t] = inv1*c2;
  }
  __syncthreads();
  if (t < 24){ int v = t/3, m = t%3; atomicAdd(&n1s[(size_t)sn*24 + t], c1s[v]*s1[m]); }
  if (t < 40){ int v = t/5, m = t%5; atomicAdd(&n2s[(size_t)sn*40 + t], c2s[v]*s2[m]); }
}

__global__ __launch_bounds__(64) void k_equi2(const float* __restrict__ w2,
    const float* __restrict__ n0m, const float* __restrict__ n1m, const float* __restrict__ n2m,
    const int* __restrict__ srcI, const int* __restrict__ dstI,
    const float* __restrict__ sh1, const float* __restrict__ sh2,
    float* __restrict__ osum, int ebase){
  int e = ebase + blockIdx.x, t = threadIdx.x;
  __shared__ float f0[64], d1[8], d2[8];
  int sn = srcI[e], dn = dstI[e];
  f0[t] = n0m[(size_t)dn*NSDIM + t];
  if (t < 8){
    float a = 0.f;
    for (int m = 0; m < 3; ++m) a += n1m[(size_t)dn*24 + t*3 + m] * sh1[(size_t)e*3 + m];
    d1[t] = a * 0.57735027f;     // 1/sqrt(3)
  } else if (t < 16){
    int v = t - 8; float a = 0.f;
    for (int m = 0; m < 5; ++m) a += n2m[(size_t)dn*40 + v*5 + m] * sh2[(size_t)e*5 + m];
    d2[v] = a * 0.44721360f;     // 1/sqrt(5)
  }
  __syncthreads();
  const float* we = w2 + (size_t)blockIdx.x*WNUMD;
  float acc = 0.f;
  for (int u = 0; u < 64; ++u) acc += we[u*64 + t] * f0[u];
  for (int v = 0; v < 8; ++v)  acc += we[4096 + v*64 + t] * d1[v];
  for (int v = 0; v < 8; ++v)  acc += we[4608 + v*64 + t] * d2[v];
  atomicAdd(&osum[(size_t)sn*NSDIM + t], acc * 0.11180340f);  // 1/sqrt(80)
}

__global__ __launch_bounds__(256) void k_out(const float* __restrict__ g,
    const float* __restrict__ Wo, const float* __restrict__ bo, float* __restrict__ out){
  __shared__ float red[256];
  int b = blockIdx.x, t = threadIdx.x;
  red[t] = g[(size_t)b*DMODEL + t] * Wo[t];
  __syncthreads();
  for (int o = 128; o > 0; o >>= 1){ if (t < o) red[t] += red[t+o]; __syncthreads(); }
  if (t == 0) out[b] = red[0] + bo[0];
}

// ---------------------------------------------------------------------------
// Host orchestration
// ---------------------------------------------------------------------------
extern "C" void kernel_launch(void* const* d_in, const int* in_sizes, int n_in,
                              void* d_out, int out_size, void* d_ws, size_t ws_size,
                              hipStream_t stream)
{
  (void)in_sizes; (void)n_in; (void)out_size; (void)ws_size;
  auto F = [&](int i){ return (const float*)d_in[i]; };
  const float* X    = F(0);
  const float* EA   = F(1);
  const int*   EI   = (const int*)d_in[2];
  const int*   BAT  = (const int*)d_in[3];
  const int* rowI = EI;            // edge_index[0]
  const int* colI = EI + NEDGES;   // edge_index[1]

  char* wsp = (char*)d_ws;
  size_t off = 0;
  auto alloc = [&](size_t bytes)->void*{
    void* p = wsp + off; off += (bytes + 255) & ~(size_t)255; return p;
  };
  auto gemm = [&](const u16* A, const u16* B, const float* bias, float* Cf, u16* Cbf,
                  int M, int K, int N, int act){
    if ((M % 128 == 0) && (K % 32 == 0)) {
      dim3 gr(N/64, M/128);
      k_wmma_gemm_ft<<<gr, 128, 0, stream>>>(A, B, bias, Cf, Cbf, M, K, N, act);
    } else {
      dim3 gr(N/64, (M + 63)/64);
      k_wmma_gemm_g<<<gr, 128, 0, stream>>>(A, B, bias, Cf, Cbf, M, K, N, act);
    }
  };
  auto tobf = [&](const float* src, size_t n)->u16*{
    u16* d = (u16*)alloc(n*2);
    int blocks = (int)((n + 255)/256); if (blocks > 4096) blocks = 4096;
    k_f32_to_bf16<<<blocks, 256, 0, stream>>>(src, d, (int)n);
    return d;
  };

  // persistent buffers
  u16*   xbf   = tobf(X, (size_t)NNODES*DIN);
  float* h     = (float*)alloc((size_t)NNODES*DMODEL*4);
  u16*   hbf   = (u16*)  alloc((size_t)NNODES*DMODEL*2);
  u16*   rbfbf = (u16*)  alloc((size_t)NEDGES*RBINS*2);
  u16*   efbf  = (u16*)  alloc((size_t)NEDGES*DMODEL*2);
  float* qv    = (float*)alloc((size_t)NNODES*DMODEL*4);
  float* kv    = (float*)alloc((size_t)NNODES*DMODEL*4);
  float* vv    = (float*)alloc((size_t)NNODES*DMODEL*4);
  float* emat  = (float*)alloc((size_t)NEDGES*DMODEL*4);
  float* alpha = (float*)alloc((size_t)NEDGES*H3*4);
  u16*   minbf = (u16*)  alloc((size_t)NEDGES*H3*2);
  float* mmu   = (float*)alloc((size_t)NEDGES*H3*4);
  u16*   msgbf = (u16*)  alloc((size_t)NEDGES*H3*2);
  float* msg2  = (float*)alloc((size_t)NEDGES*DMODEL*4);
  float* nsum  = (float*)alloc((size_t)NNODES*DMODEL*4);
  u16*   nsumbf= (u16*)  alloc((size_t)NNODES*DMODEL*2);
  float* hcat  = (float*)alloc((size_t)NNODES*DMODEL*4);
  float* stats = (float*)alloc((size_t)2*DMODEL*4);
  float* sh1   = (float*)alloc((size_t)NEDGES*3*4);
  float* sh2   = (float*)alloc((size_t)NEDGES*5*4);
  float* sbuf  = (float*)alloc((size_t)NNODES*NSDIM*4);
  u16*   r1bf  = (u16*)  alloc((size_t)NEDGES*DMODEL*2);
  float* wchnk = (float*)alloc((size_t)ECHUNK*WNUMD*4);
  float* n0s   = (float*)alloc((size_t)NNODES*NSDIM*4);
  float* n1s   = (float*)alloc((size_t)NNODES*24*4);
  float* n2s   = (float*)alloc((size_t)NNODES*40*4);
  float* cnt   = (float*)alloc((size_t)NNODES*4);
  float* osum  = (float*)alloc((size_t)NNODES*NSDIM*4);
  u16*   osumbf= (u16*)  alloc((size_t)NNODES*NSDIM*2);
  float* tmp   = (float*)alloc((size_t)NNODES*DMODEL*4);
  float* gsum  = (float*)alloc((size_t)NGRAPH*DMODEL*4);
  float* gcnt  = (float*)alloc((size_t)NGRAPH*4);
  u16*   gbf   = (u16*)  alloc((size_t)NGRAPH*DMODEL*2);
  float* gout  = (float*)alloc((size_t)NGRAPH*DMODEL*4);

  // shared weights -> bf16
  u16* Wembb = tobf(F(4),  (size_t)DIN*DMODEL);
  u16* Wrbfb = tobf(F(6),  (size_t)RBINS*DMODEL);
  u16* Wnb   = tobf(F(46), (size_t)DMODEL*NSDIM);
  u16* A1b   = tobf(F(48), (size_t)DMODEL*DMODEL);
  u16* B1b   = tobf(F(50), (size_t)DMODEL*WNUMD);
  u16* A2b   = tobf(F(52), (size_t)DMODEL*DMODEL);
  u16* B2b   = tobf(F(54), (size_t)DMODEL*WNUMD);
  u16* Wn2b  = tobf(F(56), (size_t)NSDIM*DMODEL);
  u16* Wfb   = tobf(F(58), (size_t)DMODEL*DMODEL);

  // h = x @ W_embed + b        (K=92 -> guarded kernel)
  gemm(xbf, Wembb, F(5), h, hbf, NNODES, DIN, DMODEL, 0);
  // e_feat = softplus(rbf @ W_rbf + b)   (bf16, L2-resident for 4 downstream GEMMs)
  k_rbf<<<NEDGES, 256, 0, stream>>>(EA, rbfbf);
  gemm(rbfbf, Wrbfb, F(7), nullptr, efbf, NEDGES, RBINS, DMODEL, 2);

  auto attL = [&](int pb){
    u16* Wqb  = tobf(F(pb+0),  (size_t)DMODEL*DMODEL);
    u16* Wkb  = tobf(F(pb+2),  (size_t)DMODEL*DMODEL);
    u16* Wvb  = tobf(F(pb+4),  (size_t)DMODEL*DMODEL);
    u16* Web  = tobf(F(pb+6),  (size_t)DMODEL*DMODEL);
    u16* Wcb  = tobf(F(pb+7),  (size_t)DMODEL*DMODEL);
    u16* Wmub = tobf(F(pb+9),  (size_t)H3*H3);
    u16* Wmlb = tobf(F(pb+11), (size_t)H3*DMODEL);
    gemm(hbf, Wqb, F(pb+1), qv, nullptr, NNODES, DMODEL, DMODEL, 0);
    gemm(hbf, Wkb, F(pb+3), kv, nullptr, NNODES, DMODEL, DMODEL, 0);
    gemm(hbf, Wvb, F(pb+5), vv, nullptr, NNODES, DMODEL, DMODEL, 0);
    gemm(efbf, Web, nullptr, emat, nullptr, NEDGES, DMODEL, DMODEL, 0);
    k_build_key_alpha<<<NEDGES, 256, 0, stream>>>(qv, kv, emat, rowI, colI, alpha);
    k_build_msg<<<NEDGES, 256, 0, stream>>>(vv, emat, rowI, colI, minbf);
    k_layernorm<<<NEDGES, 256, 0, stream>>>(alpha, F(pb+13), F(pb+14), H3, 1); // gate
    gemm(minbf, Wmub, F(pb+10), mmu, nullptr, NEDGES, H3, H3, 0);
    k_mul_bf<<<4096, 256, 0, stream>>>(mmu, alpha, msgbf, NEDGES*H3);
    gemm(msgbf, Wmlb, F(pb+12), msg2, nullptr, NEDGES, H3, DMODEL, 0);
    k_layernorm<<<NEDGES, 256, 0, stream>>>(msg2, F(pb+15), F(pb+16), DMODEL, 0);
    hipMemsetAsync(nsum, 0, (size_t)NNODES*DMODEL*4, stream);
    k_scatter_add<<<NEDGES, 256, 0, stream>>>(msg2, colI, nsum, DMODEL);
    k_f32_to_bf16<<<2048, 256, 0, stream>>>(nsum, nsumbf, NNODES*DMODEL);
    gemm(nsumbf, Wcb, F(pb+8), hcat, nullptr, NNODES, DMODEL, DMODEL, 0);
    k_colstats<<<DMODEL, 256, 0, stream>>>(hcat, NNODES, stats);
    k_bn_silu<<<NNODES, 256, 0, stream>>>(hcat, stats, F(pb+17), F(pb+18), h, hbf);
  };

  attL(8);   // att0

  // ---- equivariant block ----
  k_sph<<<64, 256, 0, stream>>>(EA, sh1, sh2);
  gemm(hbf, Wnb, F(47), sbuf, nullptr, NNODES, DMODEL, NSDIM, 0);
  gemm(efbf, A1b, F(49), nullptr, r1bf, NEDGES, DMODEL, DMODEL, 1);  // relu, bf16
  hipMemsetAsync(n0s, 0, (size_t)NNODES*NSDIM*4, stream);
  hipMemsetAsync(n1s, 0, (size_t)NNODES*24*4, stream);
  hipMemsetAsync(n2s, 0, (size_t)NNODES*40*4, stream);
  hipMemsetAsync(cnt, 0, (size_t)NNODES*4, stream);
  k_count<<<64, 256, 0, stream>>>(rowI, NEDGES, cnt);
  for (int c = 0; c < NEDGES/ECHUNK; ++c){     // wchunk (42MB) stays in L2
    gemm(r1bf + (size_t)c*ECHUNK*DMODEL, B1b, F(51), wchnk, nullptr,
         ECHUNK, DMODEL, WNUMD, 0);
    k_equi1<<<ECHUNK, 64, 0, stream>>>(wchnk, sbuf, rowI, colI, sh1, sh2,
                                       n0s, n1s, n2s, c*ECHUNK);
  }
  k_divmean<<<512, 256, 0, stream>>>(n0s, cnt, NNODES, NSDIM);
  k_divmean<<<256, 256, 0, stream>>>(n1s, cnt, NNODES, 24);
  k_divmean<<<256, 256, 0, stream>>>(n2s, cnt, NNODES, 40);
  gemm(efbf, A2b, F(53), nullptr, r1bf, NEDGES, DMODEL, DMODEL, 1);
  hipMemsetAsync(osum, 0, (size_t)NNODES*NSDIM*4, stream);
  for (int c = 0; c < NEDGES/ECHUNK; ++c){
    gemm(r1bf + (size_t)c*ECHUNK*DMODEL, B2b, F(55), wchnk, nullptr,
         ECHUNK, DMODEL, WNUMD, 0);
    k_equi2<<<ECHUNK, 64, 0, stream>>>(wchnk, n0s, n1s, n2s, rowI, colI,
                                       sh1, sh2, osum, c*ECHUNK);
  }
  k_divmean<<<512, 256, 0, stream>>>(osum, cnt, NNODES, NSDIM);
  k_f32_to_bf16<<<512, 256, 0, stream>>>(osum, osumbf, NNODES*NSDIM);
  gemm(osumbf, Wn2b, F(57), tmp, nullptr, NNODES, NSDIM, DMODEL, 0);
  k_add<<<2048, 256, 0, stream>>>(h, tmp, hbf, NNODES*DMODEL);

  attL(27);  // att2

  // ---- readout ----
  hipMemsetAsync(gsum, 0, (size_t)NGRAPH*DMODEL*4, stream);
  hipMemsetAsync(gcnt, 0, (size_t)NGRAPH*4, stream);
  k_scatter_add<<<NNODES, 256, 0, stream>>>(h, BAT, gsum, DMODEL);
  k_count<<<8, 256, 0, stream>>>(BAT, NNODES, gcnt);
  k_divmean<<<32, 256, 0, stream>>>(gsum, gcnt, NGRAPH, DMODEL);
  k_f32_to_bf16<<<32, 256, 0, stream>>>(gsum, gbf, NGRAPH*DMODEL);
  gemm(gbf, Wfb, F(59), gout, nullptr, NGRAPH, DMODEL, DMODEL, 3);  // silu
  k_out<<<NGRAPH, 256, 0, stream>>>(gout, F(60), F(61), (float*)d_out);
}
